// SWHeadAttention_55181739819368
// MI455X (gfx1250) — matrix operand order repaired and checked by reading, not compiled, not run on hardware
//
#include <hip/hip_runtime.h>
#include <hip/hip_bf16.h>
#include <math.h>

#define D_MODEL 1024
#define N_HEADS 16
#define DK 64
#define SEQ 2048
#define BATCH 2
#define NWIN 256
#define LN_EPS 1e-5f

typedef __attribute__((ext_vector_type(16))) _Float16 v16h;
typedef __attribute__((ext_vector_type(8)))  _Float16 v8h;
typedef __attribute__((ext_vector_type(8)))  float    v8f;
typedef __attribute__((ext_vector_type(4)))  float    v4f;

static __device__ __forceinline__ v8f wmma16(v16h a, v16h b, v8f c) {
  return __builtin_amdgcn_wmma_f32_16x16x32_f16(
      /*neg_a=*/false, a, /*neg_b=*/false, b,
      /*c_mod=*/(short)0, c, /*reuse_a=*/false, /*reuse_b=*/false);
}

// ---------------------------------------------------------------------------
// Tiled WMMA GEMM: C[M,N] = A[M,K] @ W[K,N] + bias.  f32 in/out, f16 compute.
// Block = 256 threads (8 waves). Block tile 128(M) x 64(N). K-chunks of 32.
// Double-buffered LDS (one barrier per chunk), register-staged vector loads.
// ---------------------------------------------------------------------------
__global__ __launch_bounds__(256)
void gemm_wmma_kernel(const float* __restrict__ A, const float* __restrict__ W,
                      const float* __restrict__ bias, float* __restrict__ C,
                      int M, int N, int K) {
  __shared__ _Float16 sA[2][128][32];   // 2 x 8 KB, row-major [m][k]
  __shared__ _Float16 sB[2][64][32];    // 2 x 4 KB, transposed [n][k]
  const int tid  = threadIdx.x;
  const int wave = tid >> 5;
  const int lane = tid & 31;
  const int half = lane >> 4;           // 0: lanes 0-15, 1: lanes 16-31
  const int lm   = lane & 15;
  const int m0 = blockIdx.x * 128;
  const int n0 = blockIdx.y * 64;

  // A staging: thread owns row (tid>>1), 16 contiguous k at (tid&1)*16.
  const int ar_row = tid >> 1;
  const int ar_k   = (tid & 1) * 16;
  const float* Abase = A + (size_t)(m0 + ar_row) * K + ar_k;
  // B staging: thread owns column n=(tid&63), 8 k-rows starting (tid>>6)*8.
  // Loads are coalesced across lanes (consecutive n, same k).
  const int bn  = tid & 63;
  const int bk0 = (tid >> 6) * 8;
  const float* Wbase = W + (size_t)bk0 * N + n0 + bn;

  v4f  ar[4];     // 16 floats of A tile held in registers
  float br[8];    //  8 floats of B tile held in registers

  auto loadTile = [&](int kt) {
    const v4f* ap = (const v4f*)(Abase + kt);
    #pragma unroll
    for (int i = 0; i < 4; ++i) ar[i] = ap[i];          // 4x global_load_b128
    const float* wp = Wbase + (size_t)kt * N;
    #pragma unroll
    for (int i = 0; i < 8; ++i) br[i] = wp[(size_t)i * N]; // 8x coalesced b32
  };
  auto storeTile = [&](int buf) {
    v16h ah;
    #pragma unroll
    for (int i = 0; i < 16; ++i) ah[i] = (_Float16)ar[i >> 2][i & 3];
    *(v16h*)&sA[buf][ar_row][ar_k] = ah;                // 32B LDS store
    v8h bhv;
    #pragma unroll
    for (int i = 0; i < 8; ++i) bhv[i] = (_Float16)br[i];
    *(v8h*)&sB[buf][bn][bk0] = bhv;                     // 16B LDS store
  };

  v8f acc[4];
  #pragma unroll
  for (int t = 0; t < 4; ++t) acc[t] = (v8f)(0.0f);

  const int nk = K >> 5;
  loadTile(0);
  storeTile(0);
  if (nk > 1) loadTile(32);
  __syncthreads();

  const int arow = wave * 16 + lm;
  const int kb   = half * 8;
  const int klo  = half * 16;

  for (int ki = 0; ki < nk; ++ki) {
    const int cur = ki & 1;
    if (ki + 1 < nk) storeTile(cur ^ 1);                // regs hold tile ki+1
    if (ki + 2 < nk) loadTile((ki + 2) * 32);           // prefetch tile ki+2
    // A fragment: row = wave*16+lm, K = klo..klo+15 contiguous
    v16h afr;
    #pragma unroll
    for (int j = 0; j < 8; ++j) {
      afr[j]     = sA[cur][arow][klo + j];
      afr[8 + j] = sA[cur][arow][klo + 8 + j];
    }
    // B fragments: col = t*16+lm, K = klo..klo+15 contiguous (32B ds load)
    #pragma unroll
    for (int t = 0; t < 4; ++t) {
      v16h bfr = *(const v16h*)&sB[cur][t * 16 + lm][klo];
      acc[t] = wmma16(afr, bfr, acc[t]);
    }
    __syncthreads();                                    // single barrier/chunk
  }

  // epilogue: C layout VGPR r -> row r + 8*half, col = lane%16
  #pragma unroll
  for (int t = 0; t < 4; ++t) {
    int col = n0 + t * 16 + lm;
    float bv = bias[col];
    #pragma unroll
    for (int r = 0; r < 8; ++r) {
      int row = m0 + wave * 16 + r + half * 8;
      C[(size_t)row * N + col] = acc[t][r] + bv;
    }
  }
}

// ---------------------------------------------------------------------------
// Zero-fill (attn background outside the band).
// ---------------------------------------------------------------------------
__global__ void fill_zero_kernel(float4* __restrict__ p, long long n4) {
  long long i = (long long)blockIdx.x * blockDim.x + threadIdx.x;
  long long stride = (long long)gridDim.x * blockDim.x;
  float4 z; z.x = z.y = z.z = z.w = 0.0f;
  for (; i < n4; i += stride) p[i] = z;
}

// ---------------------------------------------------------------------------
// V transpose: vbuf [B,S,H*64] -> vT [B,H,64,S]  (LDS-tiled 32x32)
// Makes context-kernel B-fragments contiguous 64B runs.
// ---------------------------------------------------------------------------
__global__ __launch_bounds__(256)
void transpose_v_kernel(const float* __restrict__ v, float* __restrict__ vT) {
  __shared__ float tile[32][33];
  const int bh = blockIdx.z;
  const int b = bh >> 4, h = bh & (N_HEADS - 1);
  const int s0 = blockIdx.x * 32;
  const int d0 = blockIdx.y * 32;
  const int tx = threadIdx.x, ty = threadIdx.y;   // (32, 8)
  #pragma unroll
  for (int i = 0; i < 4; ++i) {
    int s = s0 + ty + i * 8;
    tile[ty + i * 8][tx] = v[((size_t)(b * SEQ + s)) * D_MODEL + h * DK + d0 + tx];
  }
  __syncthreads();
  #pragma unroll
  for (int i = 0; i < 4; ++i) {
    int d = d0 + ty + i * 8;
    vT[((size_t)bh * DK + d) * SEQ + s0 + tx] = tile[tx][ty + i * 8];
  }
}

// ---------------------------------------------------------------------------
// Banded scores: one wave per (b,h, 16-row q-tile). exp(q.kT/8) for band
// tiles (masked to |i-j|<256), plus per-row sums via shfl_xor reduction.
// All fragment loads are explicit float4 (b128) reads.
// ---------------------------------------------------------------------------
__global__ __launch_bounds__(32)
void scores_kernel(const float* __restrict__ qbuf, const float* __restrict__ kbuf,
                   float* __restrict__ attn, float* __restrict__ rowsum) {
  const int id = blockIdx.x;
  const int qt = id & (SEQ / 16 - 1);
  const int bh = id >> 7;                 // SEQ/16 = 128
  const int b  = bh >> 4;
  const int h  = bh & (N_HEADS - 1);
  const int lane = threadIdx.x;
  const int half = lane >> 4, lm = lane & 15;
  const int q0 = qt * 16;
  const int kb = half * 16;
  const int klo = half * 16;

  // A fragments of q (two 32-wide K chunks over d=0..63), pre-scaled by 1/8
  const float* qrp = qbuf + ((size_t)(b * SEQ + q0 + lm)) * D_MODEL + h * DK;
  v16h aq[2];
  #pragma unroll
  for (int c = 0; c < 2; ++c) {
    const v4f* p0 = (const v4f*)(qrp + c * 32 + kb);
    const v4f* p1 = (const v4f*)(qrp + c * 32 + kb + 8);
    v4f x0 = p0[0], x1 = p0[1], y0 = p1[0], y1 = p1[1];
    #pragma unroll
    for (int w = 0; w < 4; ++w) {
      aq[c][w]      = (_Float16)(x0[w] * 0.125f);
      aq[c][4 + w]  = (_Float16)(x1[w] * 0.125f);
      aq[c][8 + w]  = (_Float16)(y0[w] * 0.125f);
      aq[c][12 + w] = (_Float16)(y1[w] * 0.125f);
    }
  }

  int jt_lo = q0 - (NWIN - 1); if (jt_lo < 0) jt_lo = 0; jt_lo >>= 4;
  int jt_hi = q0 + 15 + (NWIN - 1); if (jt_hi > SEQ - 1) jt_hi = SEQ - 1; jt_hi >>= 4;

  float rs[8];
  #pragma unroll
  for (int r = 0; r < 8; ++r) rs[r] = 0.0f;

  for (int jt = jt_lo; jt <= jt_hi; ++jt) {
    const int jrow = jt * 16 + lm;
    const float* krp = kbuf + ((size_t)(b * SEQ + jrow)) * D_MODEL + h * DK;
    v16h bkf[2];
    #pragma unroll
    for (int c = 0; c < 2; ++c) {
      const v4f* kp = (const v4f*)(krp + c * 32 + klo);
      #pragma unroll
      for (int q = 0; q < 4; ++q) {
        v4f t = kp[q];                               // global_load_b128
        #pragma unroll
        for (int w = 0; w < 4; ++w) bkf[c][q * 4 + w] = (_Float16)t[w];
      }
    }

    v8f cacc = (v8f)(0.0f);
    cacc = wmma16(aq[0], bkf[0], cacc);
    cacc = wmma16(aq[1], bkf[1], cacc);

    #pragma unroll
    for (int r = 0; r < 8; ++r) {
      int row = q0 + r + half * 8;
      int col = jt * 16 + lm;
      int d = row - col;
      float e = (d < NWIN && d > -NWIN) ? __expf(cacc[r]) : 0.0f;
      attn[((size_t)bh * SEQ + row) * SEQ + col] = e;
      rs[r] += e;
    }
  }

  // reduce each rs[r] across the 16 lanes of its half-wave
  #pragma unroll
  for (int r = 0; r < 8; ++r) {
    float v = rs[r];
    #pragma unroll
    for (int m = 1; m < 16; m <<= 1) v += __shfl_xor(v, m, 32);
    rs[r] = v;
  }
  if (lm == 0) {
    #pragma unroll
    for (int r = 0; r < 8; ++r)
      rowsum[(size_t)bh * SEQ + q0 + r + half * 8] = rs[r];
  }
}

// ---------------------------------------------------------------------------
// Band normalize: attn row /= rowsum (band columns only; rest already 0).
// ---------------------------------------------------------------------------
__global__ __launch_bounds__(64)
void norm_kernel(float* __restrict__ attn, const float* __restrict__ rowsum) {
  const int row_id = blockIdx.x;          // bh*SEQ + row
  const int row = row_id & (SEQ - 1);
  const float inv = 1.0f / rowsum[row_id];
  int jlo = row - (NWIN - 1); if (jlo < 0) jlo = 0;
  int jhi = row + (NWIN - 1); if (jhi > SEQ - 1) jhi = SEQ - 1;
  float* p = attn + (size_t)row_id * SEQ;
  for (int j = jlo + threadIdx.x; j <= jhi; j += 64) p[j] *= inv;
}

// ---------------------------------------------------------------------------
// Context: one wave per (b,h, 16-row q-tile). ctx = attn @ v over band,
// K-chunks of 32 j's. B-fragments from vT are contiguous 64B runs.
// Output written directly in [B,S,H*64] layout for the Wo GEMM.
// ---------------------------------------------------------------------------
__global__ __launch_bounds__(32)
void context_kernel(const float* __restrict__ attn, const float* __restrict__ vT,
                    float* __restrict__ ctx) {
  const int id = blockIdx.x;
  const int qt = id & (SEQ / 16 - 1);
  const int bh = id >> 7;
  const int b  = bh >> 4;
  const int h  = bh & (N_HEADS - 1);
  const int lane = threadIdx.x;
  const int half = lane >> 4, lm = lane & 15;
  const int q0 = qt * 16;
  const int kb = half * 16;

  int jlo = q0 - (NWIN - 1); if (jlo < 0) jlo = 0;
  int jhi = q0 + 15 + (NWIN - 1); if (jhi > SEQ - 1) jhi = SEQ - 1;
  const int jlo32 = jlo & ~31;
  const int jend  = (jhi >> 4) * 16;       // start of last 16-tile

  v8f acc[4];
  #pragma unroll
  for (int t = 0; t < 4; ++t) acc[t] = (v8f)(0.0f);

  const float* arow = attn + ((size_t)bh * SEQ + q0 + lm) * SEQ;

  for (int jbase = jlo32; jbase <= jend; jbase += 32) {
    // A fragment from normalized attn (values <= 1, f16-safe), two 32B runs
    const float* ap = arow + jbase;
    const v4f* a0 = (const v4f*)(ap + kb);
    const v4f* a1 = (const v4f*)(ap + kb + 8);
    v4f x0 = a0[0], x1 = a0[1], y0 = a1[0], y1 = a1[1];
    v16h afr;
    #pragma unroll
    for (int w = 0; w < 4; ++w) {
      afr[w]      = (_Float16)x0[w];
      afr[4 + w]  = (_Float16)x1[w];
      afr[8 + w]  = (_Float16)y0[w];
      afr[12 + w] = (_Float16)y1[w];
    }
    // B fragments from vT[bh][d][j]: lane owns col d, j contiguous
    const int j0 = jbase + half * 16;
    #pragma unroll
    for (int t = 0; t < 4; ++t) {
      const int d = t * 16 + lm;
      const v4f* vp = (const v4f*)(vT + ((size_t)bh * DK + d) * SEQ + j0);
      v16h bfr;
      #pragma unroll
      for (int q = 0; q < 4; ++q) {
        v4f tt = vp[q];                              // global_load_b128
        #pragma unroll
        for (int w = 0; w < 4; ++w) bfr[q * 4 + w] = (_Float16)tt[w];
      }
      acc[t] = wmma16(afr, bfr, acc[t]);
    }
  }

  #pragma unroll
  for (int t = 0; t < 4; ++t)
    #pragma unroll
    for (int r = 0; r < 8; ++r)
      ctx[((size_t)(b * SEQ + q0 + r + half * 8)) * D_MODEL + h * DK + t * 16 + lm] = acc[t][r];
}

// ---------------------------------------------------------------------------
// Residual + LayerNorm: one 256-thread block per row of 1024.
// ---------------------------------------------------------------------------
__global__ __launch_bounds__(256)
void ln_kernel(const float* __restrict__ y, const float* __restrict__ Qin,
               const float* __restrict__ gamma, const float* __restrict__ beta,
               float* __restrict__ out) {
  __shared__ float redA[8];
  __shared__ float redB[8];
  __shared__ float smu, srstd;
  const int row = blockIdx.x;
  const int t = threadIdx.x;
  const int wave = t >> 5, lane = t & 31;
  const float* yr = y + (size_t)row * D_MODEL;
  const float* qr = Qin + (size_t)row * D_MODEL;

  float x[4]; float s = 0.0f;
  #pragma unroll
  for (int i = 0; i < 4; ++i) {
    int c = t + i * 256;
    x[i] = yr[c] + qr[c];
    s += x[i];
  }
  #pragma unroll
  for (int m = 16; m >= 1; m >>= 1) s += __shfl_xor(s, m, 32);
  if (lane == 0) redA[wave] = s;
  __syncthreads();
  if (t == 0) {
    float tot = 0.0f;
    #pragma unroll
    for (int i = 0; i < 8; ++i) tot += redA[i];
    smu = tot * (1.0f / D_MODEL);
  }
  __syncthreads();
  const float mu = smu;

  float vs = 0.0f;
  #pragma unroll
  for (int i = 0; i < 4; ++i) { float d = x[i] - mu; vs += d * d; }
  #pragma unroll
  for (int m = 16; m >= 1; m >>= 1) vs += __shfl_xor(vs, m, 32);
  if (lane == 0) redB[wave] = vs;
  __syncthreads();
  if (t == 0) {
    float tot = 0.0f;
    #pragma unroll
    for (int i = 0; i < 8; ++i) tot += redB[i];
    srstd = rsqrtf(tot * (1.0f / D_MODEL) + LN_EPS);
  }
  __syncthreads();
  const float rstd = srstd;

  #pragma unroll
  for (int i = 0; i < 4; ++i) {
    int c = t + i * 256;
    out[(size_t)row * D_MODEL + c] = (x[i] - mu) * rstd * gamma[c] + beta[c];
  }
}

// ---------------------------------------------------------------------------
// Launcher. d_in order: Q,K,V,Wq,bq,Wk,bk,Wv,bv,Wo,bo,gamma,beta.
// d_out = [out (B*S*D), attn (B*H*S*S)] concatenated.
// Workspace: qbuf|kbuf|vbuf|ctx|ybuf|vT (each B*S*D f32) + rowsum (B*H*S f32)
//            = ~96.25 MB.
// ---------------------------------------------------------------------------
extern "C" void kernel_launch(void* const* d_in, const int* in_sizes, int n_in,
                              void* d_out, int out_size, void* d_ws, size_t ws_size,
                              hipStream_t stream) {
  const float* Q     = (const float*)d_in[0];
  const float* K     = (const float*)d_in[1];
  const float* V     = (const float*)d_in[2];
  const float* Wq    = (const float*)d_in[3];
  const float* bq    = (const float*)d_in[4];
  const float* Wk    = (const float*)d_in[5];
  const float* bk    = (const float*)d_in[6];
  const float* Wv    = (const float*)d_in[7];
  const float* bv    = (const float*)d_in[8];
  const float* Wo    = (const float*)d_in[9];
  const float* bo    = (const float*)d_in[10];
  const float* gamma = (const float*)d_in[11];
  const float* beta  = (const float*)d_in[12];

  const size_t NBUF = (size_t)BATCH * SEQ * D_MODEL;        // 4,194,304
  float* ws     = (float*)d_ws;
  float* qbuf   = ws;
  float* kbuf   = ws + 1 * NBUF;
  float* vbuf   = ws + 2 * NBUF;
  float* cbuf   = ws + 3 * NBUF;
  float* ybuf   = ws + 4 * NBUF;
  float* vT     = ws + 5 * NBUF;
  float* rowsum = ws + 6 * NBUF;

  float* out_final = (float*)d_out;
  float* attn      = out_final + NBUF;                       // 134,217,728 floats

  const int M = BATCH * SEQ;                                 // 4096
  dim3 ggrid(M / 128, D_MODEL / 64);                         // (32,16)

  // 1) QKV projections (WMMA GEMMs)
  gemm_wmma_kernel<<<ggrid, 256, 0, stream>>>(Q, Wq, bq, qbuf, M, D_MODEL, D_MODEL);
  gemm_wmma_kernel<<<ggrid, 256, 0, stream>>>(K, Wk, bk, kbuf, M, D_MODEL, D_MODEL);
  gemm_wmma_kernel<<<ggrid, 256, 0, stream>>>(V, Wv, bv, vbuf, M, D_MODEL, D_MODEL);

  // 2) zero attn background + transpose V for contiguous context fragments
  long long n4 = (long long)BATCH * N_HEADS * SEQ * SEQ / 4;
  fill_zero_kernel<<<4096, 256, 0, stream>>>((float4*)attn, n4);
  transpose_v_kernel<<<dim3(SEQ / 32, DK / 32, BATCH * N_HEADS), dim3(32, 8), 0, stream>>>(vbuf, vT);

  // 3) banded exp-scores + row sums (one wave per q-tile)
  const int nqt = BATCH * N_HEADS * (SEQ / 16);              // 4096
  scores_kernel<<<nqt, 32, 0, stream>>>(qbuf, kbuf, attn, rowsum);

  // 4) normalize band
  norm_kernel<<<BATCH * N_HEADS * SEQ, 64, 0, stream>>>(attn, rowsum);

  // 5) context = attn @ v  (one wave per q-tile)
  context_kernel<<<nqt, 32, 0, stream>>>(attn, vT, cbuf);

  // 6) output projection
  gemm_wmma_kernel<<<ggrid, 256, 0, stream>>>(cbuf, Wo, bo, ybuf, M, D_MODEL, D_MODEL);

  // 7) residual + LayerNorm -> final out
  ln_kernel<<<M, 256, 0, stream>>>(ybuf, Q, gamma, beta, out_final);
}
